// DeformConv_71657234366865
// MI455X (gfx1250) — matrix operand order repaired and checked by reading, not compiled, not run on hardware
//
#include <hip/hip_runtime.h>

// ---------- problem constants ----------
#define BB   8
#define CC   128
#define HH   64
#define WW   64
#define OO   128
#define KTAP 9          // 3x3
#define HWN  (HH*WW)            // 4096
#define NTOT (BB*HWN)           // 32768
#define KD   (CC*KTAP)          // 1152

typedef __attribute__((ext_vector_type(16))) __bf16 v16bf;
typedef __attribute__((ext_vector_type(8)))  float  v8f;

struct alignas(16) U4 { unsigned int a, b, c, d; };
union FragU { U4 u[2]; v16bf v; };

// RNE float -> bf16 bit pattern (stored as ushort in scratch)
__device__ __forceinline__ unsigned short f2bf(float f) {
    unsigned int u = __builtin_bit_cast(unsigned int, f);
    u = (u + 0x7FFFu + ((u >> 16) & 1u)) >> 16;
    return (unsigned short)u;
}

// CDNA5 async global->LDS copy (ASYNCcnt-tracked, no VGPR round trip)
__device__ __forceinline__ void async_load_b128(unsigned lds_off, const void* gaddr) {
    asm volatile("global_load_async_to_lds_b128 %0, %1, off"
                 :: "v"(lds_off), "v"(gaddr)
                 : "memory");
}

// ---------- 1) NCHW -> NHWC transpose of x (f32) ----------
__global__ __launch_bounds__(256)
void nchw_to_nhwc(const float* __restrict__ x, float* __restrict__ xn) {
    __shared__ float tile[32][33];
    const int b   = blockIdx.z;
    const int hw0 = blockIdx.x * 32;
    const int c0  = blockIdx.y * 32;
    const int tx  = threadIdx.x;        // 0..31
    const int ty  = threadIdx.y;        // 0..7
    #pragma unroll
    for (int i = ty; i < 32; i += 8)
        tile[i][tx] = x[((size_t)b * CC + (c0 + i)) * HWN + hw0 + tx];
    __syncthreads();
    #pragma unroll
    for (int i = ty; i < 32; i += 8)
        xn[((size_t)b * HWN + (hw0 + i)) * CC + c0 + tx] = tile[tx][i];
}

// ---------- 2) weight [O][C][3][3] f32 -> Aw[O][tap*128+c] bf16 ----------
__global__ __launch_bounds__(256)
void pack_weight(const float* __restrict__ w, unsigned short* __restrict__ Aw) {
    int idx = blockIdx.x * 256 + threadIdx.x;       // m*KD + kd
    if (idx >= OO * KD) return;
    int m   = idx / KD;
    int kd  = idx - m * KD;
    int tap = kd >> 7;          // /128
    int c   = kd & 127;
    Aw[idx] = f2bf(w[((size_t)m * CC + c) * KTAP + tap]);
}

// ---------- 3) deformable im2col: cols[n][tap*128+c] bf16 ----------
__global__ __launch_bounds__(128)
void dcn_im2col(const float* __restrict__ xn,       // NHWC
                const float* __restrict__ offset,   // [B][18][H][W]
                unsigned short* __restrict__ cols) {
    const int n   = blockIdx.x;           // 0..NTOT-1
    const int tap = blockIdx.y;           // 0..8
    const int c   = threadIdx.x;          // 0..127
    const int b   = n >> 12;
    const int hw  = n & (HWN - 1);
    const int h   = hw >> 6;
    const int w   = hw & 63;
    const int ki  = tap / 3, kj = tap - 3 * ki;

    const float* offB = offset + ((size_t)b * 2 * KTAP + 2 * tap) * HWN;
    const float dy = offB[hw];
    const float dx = offB[HWN + hw];

    const float py = (float)(h + ki - 1) + dy;   // stride=1, pad=1, dil=1
    const float px = (float)(w + kj - 1) + dx;
    const float y0f = floorf(py), x0f = floorf(px);
    const float ly = py - y0f,    lx = px - x0f;
    const int   y0 = (int)y0f,    x0 = (int)x0f;

    const float w00 = (1.f - ly) * (1.f - lx);
    const float w01 = (1.f - ly) * lx;
    const float w10 = ly * (1.f - lx);
    const float w11 = ly * lx;

    const float* xb = xn + (size_t)b * HWN * CC;
    auto samp = [&](int y, int xx) -> float {
        if ((unsigned)y >= (unsigned)HH || (unsigned)xx >= (unsigned)WW) return 0.f;
        return xb[((size_t)y * WW + xx) * CC + c];
    };
    float v = w00 * samp(y0, x0)     + w01 * samp(y0, x0 + 1)
            + w10 * samp(y0 + 1, x0) + w11 * samp(y0 + 1, x0 + 1);

    cols[(size_t)n * KD + tap * CC + c] = f2bf(v);
}

// ---------- 4) GEMM: out[b][o][hw] = Aw[o][:] . cols[n][:] + bias[o] ----------
// block: 128(M) x 128(N) tile, 8 waves, wave = 64(M) x 32(N) = 4x2 WMMA frags.
// Double-buffered LDS, tiles fetched with global_load_async_to_lds_b128 and
// retired with partial s_wait_asynccnt so tile k+1 DMA overlaps tile k WMMAs.
#define LDSPITCH 40   // 32 bf16 + 8 pad -> 80B row stride, conflict-free b128 reads
#define TILE_HWORDS (128 * LDSPITCH)      // 5120 ushorts = 10240 B per tile

__global__ __launch_bounds__(256)
void dcn_gemm(const unsigned short* __restrict__ Aw,
              const unsigned short* __restrict__ cols,
              const float* __restrict__ bias,
              float* __restrict__ out) {
    __shared__ alignas(16) unsigned short At[2][TILE_HWORDS];
    __shared__ alignas(16) unsigned short Bt[2][TILE_HWORDS];

    const int n0   = blockIdx.x * 128;
    const int t    = threadIdx.x;
    const int lane = t & 31;
    const int wave = t >> 5;
    const int wm   = wave & 1;      // M half (64 rows)
    const int wn   = wave >> 1;     // N quarter (32 cols)
    const int l16  = lane & 15;
    const int lh   = lane >> 4;     // lane half

    // LDS byte offsets of the two buffers (low 32 bits of generic LDS address)
    const unsigned baseA0 = (unsigned)(size_t)&At[0][0];
    const unsigned baseA1 = (unsigned)(size_t)&At[1][0];
    const unsigned baseB0 = (unsigned)(size_t)&Bt[0][0];
    const unsigned baseB1 = (unsigned)(size_t)&Bt[1][0];

    v8f acc[4][2];
    #pragma unroll
    for (int i = 0; i < 4; ++i)
        #pragma unroll
        for (int j = 0; j < 2; ++j)
            acc[i][j] = (v8f)(0.f);

    // issue one 128x32 A-tile + 128x32 B-tile into buffer `buf`
    // (4 async b128 per thread -> ASYNCcnt += 4 per wave, in-order completion)
    auto issue_tile = [&](int k0, int buf) {
        const unsigned bA = buf ? baseA1 : baseA0;
        const unsigned bB = buf ? baseB1 : baseB0;
        #pragma unroll
        for (int rep = 0; rep < 2; ++rep) {
            const int id  = t + rep * 256;
            const int row = id >> 2;
            const int seg = id & 3;
            const unsigned loff = (unsigned)(row * LDSPITCH + seg * 8) * 2u;
            async_load_b128(bA + loff, Aw   + (size_t)row        * KD + k0 + seg * 8);
            async_load_b128(bB + loff, cols + (size_t)(n0 + row) * KD + k0 + seg * 8);
        }
    };

    issue_tile(0, 0);
    int p = 0;

    for (int k0 = 0; k0 < KD; k0 += 32) {
        if (k0 + 32 < KD) {
            issue_tile(k0 + 32, p ^ 1);                  // prefetch next tile (DMA)
            asm volatile("s_wait_asynccnt 0x4" ::: "memory");  // retire current tile only
        } else {
            asm volatile("s_wait_asynccnt 0x0" ::: "memory");
        }
        __syncthreads();   // all waves' tile-k DMA complete -> LDS consistent

        const unsigned short* Ab = &At[p][0];
        const unsigned short* Bb = &Bt[p][0];

        // A frags: lane<16 holds K {0..7, 16..23}; lane>=16 holds K {8..15, 24..31}
        FragU a[4];
        #pragma unroll
        for (int i = 0; i < 4; ++i) {
            const int base = (wm * 64 + i * 16 + l16) * LDSPITCH + lh * 8;
            a[i].u[0] = *(const U4*)&Ab[base];
            a[i].u[1] = *(const U4*)&Ab[base + 16];
        }
        // B frags: lane<16 holds K {0..7, 16..23}; lane>=16 holds K {8..15, 24..31}
        FragU bf[2];
        #pragma unroll
        for (int j = 0; j < 2; ++j) {
            const int base = (wn * 32 + j * 16 + l16) * LDSPITCH + lh * 8;
            bf[j].u[0] = *(const U4*)&Bb[base];
            bf[j].u[1] = *(const U4*)&Bb[base + 16];
        }

        #pragma unroll
        for (int i = 0; i < 4; ++i)
            #pragma unroll
            for (int j = 0; j < 2; ++j)
                acc[i][j] = __builtin_amdgcn_wmma_f32_16x16x32_bf16(
                    false, a[i].v, false, bf[j].v,
                    (short)0, acc[i][j], false, false);

        __syncthreads();   // all reads of buffer p done before it is overwritten
        p ^= 1;
    }

    // epilogue: C/D layout -> lanes 0-15: M=r, lanes 16-31: M=r+8 (VGPR r)
    #pragma unroll
    for (int i = 0; i < 4; ++i) {
        #pragma unroll
        for (int j = 0; j < 2; ++j) {
            const int col = n0 + wn * 32 + j * 16 + l16;   // global n
            const int b   = col >> 12;
            const int hw  = col & (HWN - 1);
            #pragma unroll
            for (int r = 0; r < 8; ++r) {
                const int m = wm * 64 + i * 16 + lh * 8 + r;
                out[(((size_t)b * OO + m) << 12) + hw] = acc[i][j][r] + bias[m];
            }
        }
    }
}

// ---------- launcher ----------
extern "C" void kernel_launch(void* const* d_in, const int* in_sizes, int n_in,
                              void* d_out, int out_size, void* d_ws, size_t ws_size,
                              hipStream_t stream) {
    const float* x      = (const float*)d_in[0];   // [8,128,64,64]
    const float* offset = (const float*)d_in[1];   // [8,18,64,64]
    const float* weight = (const float*)d_in[2];   // [128,128,3,3]
    const float* bias   = (const float*)d_in[3];   // [128]
    float*       out    = (float*)d_out;           // [8,128,64,64]

    char* ws = (char*)d_ws;
    float*          xn   = (float*)ws;                                   // 16,777,216 B
    unsigned short* cols = (unsigned short*)(ws + (size_t)16777216);     // 75,497,472 B
    unsigned short* Aw   = (unsigned short*)(ws + (size_t)16777216 + 75497472); // 294,912 B

    nchw_to_nhwc<<<dim3(HWN / 32, CC / 32, BB), dim3(32, 8), 0, stream>>>(x, xn);
    pack_weight<<<(OO * KD + 255) / 256, 256, 0, stream>>>(weight, Aw);
    dcn_im2col<<<dim3(NTOT, KTAP), 128, 0, stream>>>(xn, offset, cols);
    dcn_gemm<<<NTOT / 128, 256, 0, stream>>>(Aw, cols, bias, out);
}
